// diff_attention_28209345200667
// MI455X (gfx1250) — matrix operand order repaired and checked by reading, not compiled, not run on hardware
//
#include <hip/hip_runtime.h>

// Differential attention, B=8, S=2048, E=1024, MI455X (gfx1250, wave32, WMMA).
//
// Math reductions vs reference:
//  - Q1 rows identical per batch -> softmax(A1) is one vector p1[b,:].
//  - q1.K1[b,k] = X1h[b,k,:] . (Wk1 @ q1[b])  -> K1 GEMM collapses to GEMV u.
//  - bk1 shifts whole softmax rows -> cancels.
//  - V = x1v(broadcast) + Vbot;  softmax rows sum to 1 ->
//    out = (1-a)*x1v + p1@Vbot - a*(softmax(A2)@Vbot).
// Heavy GEMMs (projections, scores, PV) run as bf16 WMMA with f32 accumulate.
// Scores/PV kernels double-buffer LDS tiles via GLOBAL_LOAD_ASYNC_TO_LDS_B128
// (ASYNCcnt-tracked) when the toolchain exposes the builtin.

#define B_ 8
#define S_ 2048
#define E_ 1024
#define LDP 40  // 32 K-elems + 8 bf16 pad -> 80B row stride, conflict-free

typedef __bf16 bf16;
typedef __attribute__((ext_vector_type(16))) __bf16 v16bf;
typedef __attribute__((ext_vector_type(8)))  __bf16 v8bf;
typedef __attribute__((ext_vector_type(8)))  float  v8f;
typedef __attribute__((ext_vector_type(4)))  float  v4f;
typedef int v4i __attribute__((vector_size(16)));
typedef __attribute__((address_space(1))) v4i* gptr128;
typedef __attribute__((address_space(3))) v4i* lptr128;

#if defined(__has_builtin)
#if __has_builtin(__builtin_amdgcn_global_load_async_to_lds_b128) && \
    __has_builtin(__builtin_amdgcn_s_wait_asynccnt)
#define USE_ASYNC 1
#endif
#endif
#ifndef USE_ASYNC
#define USE_ASYNC 0
#endif

__device__ inline v8f wmma_bf16(v16bf a, v16bf b, v8f c) {
  return __builtin_amdgcn_wmma_f32_16x16x32_bf16(false, a, false, b, (short)0, c,
                                                 false, false);
}

__device__ inline v16bf mk16(v8bf lo, v8bf hi) {
  v16bf r;
#pragma unroll
  for (int i = 0; i < 8; ++i) { r[i] = lo[i]; r[i + 8] = hi[i]; }
  return r;
}
// A operand 16x32: lane m(+16h) holds row m; elems0-7 = K 8h.., elems8-15 = K 16+8h..
__device__ inline v16bf frag_a(const bf16* row, int h) {
  return mk16(*(const v8bf*)(row + 8 * h), *(const v8bf*)(row + 16 + 8 * h));
}
// B operand 32x16: lane n(+16h) holds col n; elems0-7 = K 8h.., elems8-15 = K 16+8h..
__device__ inline v16bf frag_b(const bf16* row, int h) {
  return mk16(*(const v8bf*)(row + 8 * h), *(const v8bf*)(row + 16 + 8 * h));
}

// 16-byte global->LDS tile stage: async DMA to LDS when available.
__device__ inline void stage16(const bf16* g, bf16* l) {
#if USE_ASYNC
  __builtin_amdgcn_global_load_async_to_lds_b128(
      (gptr128)((__attribute__((address_space(1))) void*)g),
      (lptr128)((__attribute__((address_space(3))) void*)l), 0, 0);
#else
  *(v8bf*)l = *(const v8bf*)g;
#endif
}
__device__ inline void wait_async_pending() {  // current tile done, next in flight
#if USE_ASYNC
  __builtin_amdgcn_s_wait_asynccnt(4);
#endif
}
__device__ inline void wait_async_zero() {
#if USE_ASYNC
  __builtin_amdgcn_s_wait_asynccnt(0);
#endif
}

__device__ inline float block_red(float v, float* red, bool domax) {
  const int tid = threadIdx.x;
  __syncthreads();
  red[tid] = v;
  __syncthreads();
#pragma unroll
  for (int s = 128; s > 0; s >>= 1) {
    if (tid < s) red[tid] = domax ? fmaxf(red[tid], red[tid + s]) : (red[tid] + red[tid + s]);
    __syncthreads();
  }
  return red[0];
}

// ---- tiny per-batch vectors: q1 = X1@Wq1 + bq1 ; x1v = X1@Wv_top + bv -------
__global__ __launch_bounds__(256) void vec_proj_kernel(
    const float* __restrict__ X1, const float* __restrict__ Wq1,
    const float* __restrict__ bq1, const float* __restrict__ Wv,
    const float* __restrict__ bv, float* __restrict__ q1, float* __restrict__ x1v) {
  const int tid = blockIdx.x * 256 + threadIdx.x;  // 2*B*E threads
  const int half = tid / (B_ * E_);
  const int r = tid - half * (B_ * E_);
  const int b = r / E_, j = r - b * E_;
  const float* x = X1 + (size_t)b * E_;
  const float* W = (half == 0) ? Wq1 : Wv;  // Wv top half rows 0..E-1
  float s = 0.f;
  for (int e = 0; e < E_; ++e) s += x[e] * W[(size_t)e * E_ + j];
  if (half == 0) q1[r] = s + bq1[j];
  else           x1v[r] = s + bv[j];
}

// ---- u[b,e'] = sum_j Wk1[e',j] * q1[b,j] ------------------------------------
__global__ __launch_bounds__(256) void u_kernel(const float* __restrict__ Wk1,
                                                const float* __restrict__ q1,
                                                float* __restrict__ u) {
  __shared__ float red[256];
  const int o = blockIdx.x, tid = threadIdx.x;  // B*E blocks
  const int b = o / E_, ep = o - b * E_;
  const float* wr = Wk1 + (size_t)ep * E_;
  const float* qb = q1 + (size_t)b * E_;
  float s = 0.f;
#pragma unroll
  for (int i = 0; i < 4; ++i) { const int j = tid + 256 * i; s += wr[j] * qb[j]; }
  s = block_red(s, red, false);
  if (tid == 0) u[o] = s;
}

// ---- d1[b,k] = (X1h[b,k,:] . u[b,:]) / sqrt(E) ------------------------------
__global__ __launch_bounds__(256) void d1_kernel(const float* __restrict__ X1h,
                                                 const float* __restrict__ u,
                                                 float* __restrict__ d1) {
  __shared__ float red[256];
  const int o = blockIdx.x, tid = threadIdx.x;  // B*S blocks
  const int b = o / S_;
  const float* xr = X1h + (size_t)o * E_;
  const float* ub = u + (size_t)b * E_;
  float s = 0.f;
#pragma unroll
  for (int i = 0; i < 4; ++i) { const int e = tid + 256 * i; s += xr[e] * ub[e]; }
  s = block_red(s, red, false);
  if (tid == 0) d1[o] = s * 0.03125f;  // 1/sqrt(1024)
}

// ---- p1 = softmax(d1) per batch row of S ------------------------------------
__global__ __launch_bounds__(256) void p1_kernel(const float* __restrict__ d1,
                                                 float* __restrict__ p1) {
  __shared__ float red[256];
  const int tid = threadIdx.x;
  const float* dr = d1 + (size_t)blockIdx.x * S_;
  float* pr = p1 + (size_t)blockIdx.x * S_;
  float v[8], m = -3.0e38f;
#pragma unroll
  for (int i = 0; i < 8; ++i) { v[i] = dr[tid + 256 * i]; m = fmaxf(m, v[i]); }
  m = block_red(m, red, true);
  float s = 0.f;
#pragma unroll
  for (int i = 0; i < 8; ++i) { v[i] = __expf(v[i] - m); s += v[i]; }
  s = block_red(s, red, false);
  const float inv = 1.0f / s;
#pragma unroll
  for (int i = 0; i < 8; ++i) pr[tid + 256 * i] = v[i] * inv;
}

// ---- fused projection GEMM: Q2,K2 row-major bf16; Vbot stored transposed ----
__global__ __launch_bounds__(256) void proj_kernel(
    const float* __restrict__ X2, const float* __restrict__ Wq2,
    const float* __restrict__ bq2, const float* __restrict__ Wk2,
    const float* __restrict__ bk2, const float* __restrict__ Wv,
    bf16* __restrict__ Q2, bf16* __restrict__ K2, bf16* __restrict__ VT) {
  __shared__ bf16 lA[128 * LDP];
  __shared__ bf16 lW[3 * 128 * LDP];
  const int tid = threadIdx.x;
  const int m0 = blockIdx.x * 128;  // rows of flattened [B*S]
  const int n0 = blockIdx.y * 128;  // output feature tile
  const int wave = tid >> 5, lane = tid & 31, h = lane >> 4, l = lane & 15;
  const float* Wp[3] = {Wq2, Wk2, Wv + (size_t)E_ * E_};
  v8f acc[3][8] = {};
  const int arow = tid >> 1, ahalf = tid & 1;
  const int wk = tid >> 3, wg = tid & 7;
  for (int k0 = 0; k0 < E_; k0 += 32) {
    {  // stage X2 tile (fp32 -> bf16), row-major [128][32]
      const float* src = X2 + (size_t)(m0 + arow) * E_ + k0 + 16 * ahalf;
      bf16* dst = lA + arow * LDP + 16 * ahalf;
#pragma unroll
      for (int i = 0; i < 16; i += 4) {
        v4f v = *(const v4f*)(src + i);
#pragma unroll
        for (int c = 0; c < 4; ++c) dst[i + c] = (bf16)v[c];
      }
    }
#pragma unroll
    for (int wt = 0; wt < 3; ++wt) {  // stage W tiles transposed: lW[n][k]
      const float* src = Wp[wt] + (size_t)(k0 + wk) * E_ + n0 + 16 * wg;
      bf16* dst = lW + wt * 128 * LDP;
#pragma unroll
      for (int i = 0; i < 16; i += 4) {
        v4f v = *(const v4f*)(src + i);
#pragma unroll
        for (int c = 0; c < 4; ++c) dst[(16 * wg + i + c) * LDP + wk] = (bf16)v[c];
      }
    }
    __syncthreads();
    const v16bf af = frag_a(lA + (16 * wave + l) * LDP, h);
#pragma unroll
    for (int wt = 0; wt < 3; ++wt)
#pragma unroll
      for (int j = 0; j < 8; ++j) {
        const v16bf bfv = frag_b(lW + wt * 128 * LDP + (16 * j + l) * LDP, h);
        acc[wt][j] = wmma_bf16(af, bfv, acc[wt][j]);
      }
    __syncthreads();
  }
  const int b = m0 / S_;
  const int srow0 = (m0 % S_) + 16 * wave + 8 * h;
#pragma unroll
  for (int j = 0; j < 8; ++j) {
    const int col = n0 + 16 * j + l;
    const float biq = bq2[col], bik = bk2[col];
#pragma unroll
    for (int r = 0; r < 8; ++r) {
      const int row = m0 + 16 * wave + 8 * h + r;
      Q2[(size_t)row * E_ + col] = (bf16)(acc[0][j][r] + biq);
      K2[(size_t)row * E_ + col] = (bf16)(acc[1][j][r] + bik);
    }
    v8bf pk;  // VT[b][e][s]: 8 consecutive s per lane -> packed 16B store
#pragma unroll
    for (int r = 0; r < 8; ++r) pk[r] = (bf16)acc[2][j][r];
    *(v8bf*)(VT + ((size_t)b * E_ + col) * S_ + srow0) = pk;
  }
}

// ---- base[b,e] = (1-alpha)*x1v + p1 @ Vbot ----------------------------------
__global__ __launch_bounds__(256) void base_kernel(const bf16* __restrict__ VT,
                                                   const float* __restrict__ p1,
                                                   const float* __restrict__ x1v,
                                                   const float* __restrict__ alpha,
                                                   float* __restrict__ base) {
  __shared__ float red[256];
  const int o = blockIdx.x, tid = threadIdx.x;  // B*E blocks
  const int b = o / E_, e = o - b * E_;
  const bf16* vr = VT + ((size_t)b * E_ + e) * S_;
  const float* pb = p1 + (size_t)b * S_;
  float s = 0.f;
#pragma unroll
  for (int i = 0; i < 8; ++i) { const int k = tid + 256 * i; s += (float)vr[k] * pb[k]; }
  s = block_red(s, red, false);
  if (tid == 0) base[o] = (1.0f - alpha[0]) * x1v[o] + s;
}

// ---- scores: A2 = (Q2 @ K2^T) / sqrt(E), bf16; async double-buffered --------
__global__ __launch_bounds__(256) void scores_kernel(const bf16* __restrict__ Q2,
                                                     const bf16* __restrict__ K2,
                                                     bf16* __restrict__ A2) {
  __shared__ bf16 lQ[2][128 * LDP];
  __shared__ bf16 lK[2][128 * LDP];
  const int tid = threadIdx.x;
  const int b = blockIdx.z;
  const int m0 = blockIdx.x * 128;  // q tile
  const int n0 = blockIdx.y * 128;  // k tile
  const int wave = tid >> 5, lane = tid & 31, h = lane >> 4, l = lane & 15;
  v8f acc[8] = {};
  const int row = tid >> 1, half = tid & 1;
  const bf16* qbase = Q2 + ((size_t)b * S_ + m0 + row) * E_ + 16 * half;
  const bf16* kbase = K2 + ((size_t)b * S_ + n0 + row) * E_ + 16 * half;
  const int loff = row * LDP + 16 * half;
  stage16(qbase, &lQ[0][loff]);
  stage16(qbase + 8, &lQ[0][loff + 8]);
  stage16(kbase, &lK[0][loff]);
  stage16(kbase + 8, &lK[0][loff + 8]);
  const int NT = E_ / 32;
  for (int kt = 0; kt < NT; ++kt) {
    const int cur = kt & 1;
    if (kt + 1 < NT) {  // prefetch next tile into the other buffer
      const bf16* q = qbase + 32 * (kt + 1);
      const bf16* k = kbase + 32 * (kt + 1);
      stage16(q, &lQ[cur ^ 1][loff]);
      stage16(q + 8, &lQ[cur ^ 1][loff + 8]);
      stage16(k, &lK[cur ^ 1][loff]);
      stage16(k + 8, &lK[cur ^ 1][loff + 8]);
      wait_async_pending();
    } else {
      wait_async_zero();
    }
    __syncthreads();
    const v16bf af = frag_a(&lQ[cur][(16 * wave + l) * LDP], h);
    v16bf bf8[8];
#pragma unroll
    for (int j = 0; j < 8; ++j) bf8[j] = frag_b(&lK[cur][(16 * j + l) * LDP], h);
#pragma unroll
    for (int j = 0; j < 8; ++j) acc[j] = wmma_bf16(af, bf8[j], acc[j]);
    __syncthreads();
  }
  const float sc = 0.03125f;
#pragma unroll
  for (int j = 0; j < 8; ++j) {
    const int kc = n0 + 16 * j + l;
#pragma unroll
    for (int r = 0; r < 8; ++r) {
      const int q = m0 + 16 * wave + 8 * h + r;
      A2[((size_t)b * S_ + q) * S_ + kc] = (bf16)(acc[j][r] * sc);
    }
  }
}

// ---- in-place row softmax over A2 (rows of length S) ------------------------
__global__ __launch_bounds__(256) void softmax_kernel(bf16* __restrict__ A) {
  __shared__ float red[256];
  const int tid = threadIdx.x;
  bf16* row = A + (size_t)blockIdx.x * S_;
  float v[8], m = -3.0e38f;
#pragma unroll
  for (int i = 0; i < 8; ++i) { v[i] = (float)row[tid + 256 * i]; m = fmaxf(m, v[i]); }
  m = block_red(m, red, true);
  float s = 0.f;
#pragma unroll
  for (int i = 0; i < 8; ++i) { v[i] = __expf(v[i] - m); s += v[i]; }
  s = block_red(s, red, false);
  const float inv = 1.0f / s;
#pragma unroll
  for (int i = 0; i < 8; ++i) row[tid + 256 * i] = (bf16)(v[i] * inv);
}

// ---- out = base - alpha * (P @ Vbot); async double-buffered -----------------
__global__ __launch_bounds__(256) void pv_kernel(const bf16* __restrict__ P,
                                                 const bf16* __restrict__ VT,
                                                 const float* __restrict__ base,
                                                 const float* __restrict__ alpha,
                                                 float* __restrict__ out) {
  __shared__ bf16 lP[2][128 * LDP];
  __shared__ bf16 lV[2][128 * LDP];
  const int tid = threadIdx.x;
  const int b = blockIdx.z;
  const int m0 = blockIdx.x * 128;  // q tile
  const int n0 = blockIdx.y * 128;  // e tile
  const int wave = tid >> 5, lane = tid & 31, h = lane >> 4, l = lane & 15;
  v8f acc[8] = {};
  const int row = tid >> 1, half = tid & 1;
  const bf16* pbase = P + ((size_t)b * S_ + m0 + row) * S_ + 16 * half;
  const bf16* vbase = VT + ((size_t)b * E_ + n0 + row) * S_ + 16 * half;
  const int loff = row * LDP + 16 * half;
  stage16(pbase, &lP[0][loff]);
  stage16(pbase + 8, &lP[0][loff + 8]);
  stage16(vbase, &lV[0][loff]);
  stage16(vbase + 8, &lV[0][loff + 8]);
  const int NT = S_ / 32;
  for (int kt = 0; kt < NT; ++kt) {
    const int cur = kt & 1;
    if (kt + 1 < NT) {
      const bf16* p = pbase + 32 * (kt + 1);
      const bf16* v = vbase + 32 * (kt + 1);
      stage16(p, &lP[cur ^ 1][loff]);
      stage16(p + 8, &lP[cur ^ 1][loff + 8]);
      stage16(v, &lV[cur ^ 1][loff]);
      stage16(v + 8, &lV[cur ^ 1][loff + 8]);
      wait_async_pending();
    } else {
      wait_async_zero();
    }
    __syncthreads();
    const v16bf af = frag_a(&lP[cur][(16 * wave + l) * LDP], h);
    v16bf bf8[8];
#pragma unroll
    for (int j = 0; j < 8; ++j) bf8[j] = frag_b(&lV[cur][(16 * j + l) * LDP], h);
#pragma unroll
    for (int j = 0; j < 8; ++j) acc[j] = wmma_bf16(af, bf8[j], acc[j]);
    __syncthreads();
  }
  const float al = alpha[0];
#pragma unroll
  for (int j = 0; j < 8; ++j) {
    const int e = n0 + 16 * j + l;
    const float bs = base[(size_t)b * E_ + e];
#pragma unroll
    for (int r = 0; r < 8; ++r) {
      const int q = m0 + 16 * wave + 8 * h + r;
      out[((size_t)b * S_ + q) * E_ + e] = bs - al * acc[j][r];
    }
  }
}

extern "C" void kernel_launch(void* const* d_in, const int* in_sizes, int n_in,
                              void* d_out, int out_size, void* d_ws, size_t ws_size,
                              hipStream_t stream) {
  (void)in_sizes; (void)n_in; (void)out_size; (void)ws_size;
  const float* X1  = (const float*)d_in[0];
  const float* X2  = (const float*)d_in[1];
  const float* X1h = (const float*)d_in[2];
  const float* Wq1 = (const float*)d_in[3];
  const float* bq1 = (const float*)d_in[4];
  const float* Wq2 = (const float*)d_in[5];
  const float* bq2 = (const float*)d_in[6];
  const float* Wk1 = (const float*)d_in[7];
  // d_in[8] = bk1: adds per-batch constant to every A1 row -> cancels in softmax.
  const float* Wk2 = (const float*)d_in[9];
  const float* bk2 = (const float*)d_in[10];
  const float* Wv  = (const float*)d_in[11];
  const float* bv  = (const float*)d_in[12];
  const float* alpha = (const float*)d_in[13];
  float* out = (float*)d_out;

  char* ws = (char*)d_ws;
  size_t off = 0;
  auto carve = [&](size_t bytes) -> char* {
    char* p = ws + off;
    off += (bytes + 255) & ~(size_t)255;
    return p;
  };
  bf16* Q2  = (bf16*)carve((size_t)B_ * S_ * E_ * 2);  // 32 MB
  bf16* K2  = (bf16*)carve((size_t)B_ * S_ * E_ * 2);  // 32 MB
  bf16* VT  = (bf16*)carve((size_t)B_ * E_ * S_ * 2);  // 32 MB  [b][e][s]
  bf16* P   = (bf16*)carve((size_t)B_ * S_ * S_ * 2);  // 64 MB  scores->softmax in place
  float* q1   = (float*)carve((size_t)B_ * E_ * 4);
  float* x1v  = (float*)carve((size_t)B_ * E_ * 4);
  float* u    = (float*)carve((size_t)B_ * E_ * 4);
  float* d1   = (float*)carve((size_t)B_ * S_ * 4);
  float* p1   = (float*)carve((size_t)B_ * S_ * 4);
  float* base = (float*)carve((size_t)B_ * E_ * 4);

  vec_proj_kernel<<<(2 * B_ * E_) / 256, 256, 0, stream>>>(X1, Wq1, bq1, Wv, bv, q1, x1v);
  u_kernel<<<B_ * E_, 256, 0, stream>>>(Wk1, q1, u);
  d1_kernel<<<B_ * S_, 256, 0, stream>>>(X1h, u, d1);
  p1_kernel<<<B_, 256, 0, stream>>>(d1, p1);
  proj_kernel<<<dim3((B_ * S_) / 128, E_ / 128), 256, 0, stream>>>(X2, Wq2, bq2, Wk2,
                                                                   bk2, Wv, Q2, K2, VT);
  base_kernel<<<B_ * E_, 256, 0, stream>>>(VT, p1, x1v, alpha, base);
  scores_kernel<<<dim3(S_ / 128, S_ / 128, B_), 256, 0, stream>>>(Q2, K2, P);
  softmax_kernel<<<B_ * S_, 256, 0, stream>>>(P);
  pv_kernel<<<dim3(S_ / 128, E_ / 128, B_), 256, 0, stream>>>(P, VT, base, alpha, out);
}
